// AttentiveFP_3238405341434
// MI455X (gfx1250) — compile-verified, matrix operand order checked by reading, not run on hardware
//
#include <hip/hip_runtime.h>
#include <math.h>

// ---------------------------------------------------------------------------
// Problem dims (AttentiveFP)
// ---------------------------------------------------------------------------
#define NN    50000
#define EE    1000000
#define HH    64
#define EDD   16
#define GG    512
#define IN_CC 39
#define OUT_CC 128
#define SLOPE 0.01f

typedef __attribute__((ext_vector_type(16))) _Float16 v16h;
typedef __attribute__((ext_vector_type(8)))  float    v8f;

__device__ __forceinline__ float leakyf(float v) { return v > 0.f ? v : SLOPE * v; }

__device__ __forceinline__ void pack16(v16h& v, const float4 p0, const float4 p1,
                                       const float4 p2, const float4 p3)
{
    v[0]  = (_Float16)p0.x; v[1]  = (_Float16)p0.y;
    v[2]  = (_Float16)p0.z; v[3]  = (_Float16)p0.w;
    v[4]  = (_Float16)p1.x; v[5]  = (_Float16)p1.y;
    v[6]  = (_Float16)p1.z; v[7]  = (_Float16)p1.w;
    v[8]  = (_Float16)p2.x; v[9]  = (_Float16)p2.y;
    v[10] = (_Float16)p2.z; v[11] = (_Float16)p2.w;
    v[12] = (_Float16)p3.x; v[13] = (_Float16)p3.y;
    v[14] = (_Float16)p3.z; v[15] = (_Float16)p3.w;
}

// ---------------------------------------------------------------------------
// WMMA GEMM, K fixed at 64:  C[M,F] = A[M,64] @ W[F,ldw]^T (cols 0..63) (+bias)
// Requirements (all satisfied statically by this model):
//   M % 16 == 0  (N=50000=3125*16, G=512)   -> no row guards, EXEC stays full
//   F % 64 == 0  (64 / 128 / 192)
// Block = 128 threads = 4 waves; one 16-row stripe of A shared by 4 waves,
// each wave owns one 16-col tile. Fragment loads are float4 (global_load_b128):
// per lane the 16 half-elements are two contiguous 8-float runs
//   K = k0 + 8*half .. +7   and   K = k0 + 16 + 8*half .. +7      (ISA 7.12.2)
// ---------------------------------------------------------------------------
__global__ void gemm_nt_wmma(const float* __restrict__ A,
                             const float* __restrict__ W, int ldw,
                             const float* __restrict__ bias,
                             float* __restrict__ C, int ldc)
{
    const int lane = threadIdx.x & 31;
    const int wave = threadIdx.x >> 5;                 // 0..3
    const int m0   = blockIdx.x * 16;
    const int n0   = blockIdx.y * 64 + wave * 16;
    const int half = lane >> 4;                        // 0 or 1
    const int lidx = lane & 15;

    const float* Ar = A + (size_t)(m0 + lidx) * HH + half * 8;
    const float* Wr = W + (size_t)(n0 + lidx) * ldw + half * 8;

    v8f c = {};
#pragma unroll
    for (int k0 = 0; k0 < HH; k0 += 32) {
        const float4 a0 = *(const float4*)(Ar + k0);
        const float4 a1 = *(const float4*)(Ar + k0 + 4);
        const float4 a2 = *(const float4*)(Ar + k0 + 16);
        const float4 a3 = *(const float4*)(Ar + k0 + 20);
        const float4 b0 = *(const float4*)(Wr + k0);
        const float4 b1 = *(const float4*)(Wr + k0 + 4);
        const float4 b2 = *(const float4*)(Wr + k0 + 16);
        const float4 b3 = *(const float4*)(Wr + k0 + 20);
        v16h a, b;
        pack16(a, a0, a1, a2, a3);
        pack16(b, b0, b1, b2, b3);
        c = __builtin_amdgcn_wmma_f32_16x16x32_f16(
                /*neg_a=*/false, a, /*neg_b=*/false, b,
                /*c_mod=*/(short)0, c, /*reuse_a=*/false, /*reuse_b=*/false);
    }

    const int col = n0 + lidx;
    const float bv = bias ? bias[col] : 0.0f;
#pragma unroll
    for (int e = 0; e < 8; ++e) {
        const int row = m0 + e + half * 8;
        C[(size_t)row * ldc + col] = c[e] + bv;
    }
}

// ---------------------------------------------------------------------------
// lin1: x0 = leaky_relu(x @ lin1_w.T + b), K = 39 (scalar: K not mult of 32)
// ---------------------------------------------------------------------------
__global__ void lin1_k(const float* __restrict__ x, const float* __restrict__ w,
                       const float* __restrict__ b, float* __restrict__ o)
{
    int t = blockIdx.x * blockDim.x + threadIdx.x;
    if (t >= NN * HH) return;
    int i = t >> 6, j = t & 63;
    float acc = b[j];
    const float* xr = x + (size_t)i * IN_CC;
    const float* wr = w + (size_t)j * IN_CC;
#pragma unroll
    for (int k = 0; k < IN_CC; ++k) acc += xr[k] * wr[k];
    o[t] = leakyf(acc);
}

// r[i] = dot(A[i,0:64], v)
__global__ void rowdot_k(const float* __restrict__ A, const float* __restrict__ v,
                         float* __restrict__ r, int M)
{
    int i = blockIdx.x * blockDim.x + threadIdx.x;
    if (i >= M) return;
    const float* a = A + (size_t)i * HH;
    float acc = 0.f;
#pragma unroll
    for (int j = 0; j < HH; ++j) acc += a[j] * v[j];
    r[i] = acc;
}

// ---------------------------------------------------------------------------
// GATEConv edge logits:
//   t   = leaky(tA[src] + edge_attr @ W1b.T)      (rank-16 per-edge GEMM)
//   er  = leaky(dot(t, att_l) + rdst[dst])
// ---------------------------------------------------------------------------
__global__ void gate_edge_logits_k(const float* __restrict__ eattr,
                                   const int* __restrict__ src,
                                   const int* __restrict__ dst,
                                   const float* __restrict__ tA,
                                   const float* __restrict__ glin1, // [64 x 80]
                                   const float* __restrict__ att_l,
                                   const float* __restrict__ rdst,
                                   float* __restrict__ er)
{
    __shared__ float w1b[HH * EDD];
    __shared__ float al[HH];
    for (int t = threadIdx.x; t < HH * EDD; t += blockDim.x)
        w1b[t] = glin1[(size_t)(t >> 4) * (HH + EDD) + HH + (t & 15)];
    for (int t = threadIdx.x; t < HH; t += blockDim.x) al[t] = att_l[t];
    __syncthreads();

    int e = blockIdx.x * blockDim.x + threadIdx.x;
    if (e >= EE) return;
    float ea[EDD];
#pragma unroll
    for (int k = 0; k < EDD; ++k) ea[k] = eattr[(size_t)e * EDD + k];
    const float* ta = tA + (size_t)src[e] * HH;
    __builtin_prefetch(ta, 0, 0);               // global_prefetch_b8
    float acc = 0.f;
    for (int n = 0; n < HH; ++n) {
        float tb = 0.f;
#pragma unroll
        for (int k = 0; k < EDD; ++k) tb += ea[k] * w1b[n * EDD + k];
        acc += leakyf(ta[n] + tb) * al[n];
    }
    er[e] = leakyf(acc + rdst[dst[e]]);
}

// GATConv edge logits: er = leaky(asrc[src] + adst[dst])
__global__ void edge_logits_k(const float* __restrict__ asrc,
                              const float* __restrict__ adst,
                              const int* __restrict__ src,
                              const int* __restrict__ dst,
                              float* __restrict__ er)
{
    int e = blockIdx.x * blockDim.x + threadIdx.x;
    if (e >= EE) return;
    er[e] = leakyf(asrc[src[e]] + adst[dst[e]]);
}

// mol logits: er[n] = leaky(rs[n] + dd[batch[n]])
__global__ void mol_logits_k(const float* __restrict__ rs, const float* __restrict__ dd,
                             const int* __restrict__ batch, float* __restrict__ er)
{
    int i = blockIdx.x * blockDim.x + threadIdx.x;
    if (i >= NN) return;
    er[i] = leakyf(rs[i] + dd[batch[i]]);
}

// ---------------------------------------------------------------------------
// Segment softmax machinery (float atomic max via order-preserving uint key)
// ---------------------------------------------------------------------------
#define ENC_NEG_INF 0x007FFFFFu   // encode(-inf)

__global__ void fill_u32_k(unsigned* __restrict__ p, unsigned v, int n)
{
    int i = blockIdx.x * blockDim.x + threadIdx.x;
    if (i < n) p[i] = v;
}

__global__ void seg_max_k(const float* __restrict__ a, const int* __restrict__ seg,
                          int n, unsigned* __restrict__ m)
{
    int i = blockIdx.x * blockDim.x + threadIdx.x;
    if (i >= n) return;
    unsigned u   = __float_as_uint(a[i]);
    unsigned key = (u & 0x80000000u) ? ~u : (u | 0x80000000u);
    atomicMax(&m[seg[i]], key);
}

__global__ void seg_max_fin_k(unsigned* __restrict__ m, int n)
{
    int i = blockIdx.x * blockDim.x + threadIdx.x;
    if (i >= n) return;
    unsigned key = m[i];
    float f = (key & 0x80000000u) ? __uint_as_float(key & 0x7FFFFFFFu)
                                  : __uint_as_float(~key);
    if (!(f >= -3.0e38f && f <= 3.0e38f)) f = 0.f;   // non-finite -> 0
    ((float*)m)[i] = f;
}

__global__ void seg_expsum_k(float* __restrict__ a, const int* __restrict__ seg,
                             int n, const float* __restrict__ m, float* __restrict__ s)
{
    int i = blockIdx.x * blockDim.x + threadIdx.x;
    if (i >= n) return;
    int g = seg[i];
    float e = __expf(a[i] - m[g]);
    a[i] = e;
    atomicAdd(&s[g], e);
}

// out[seg[i], :] += vec[srcidx ? srcidx[i] : i, :] * (a[i] / (s[seg[i]] + eps))
__global__ void seg_scatter_k(const float* __restrict__ a, const float* __restrict__ s,
                              const int* __restrict__ seg, const int* __restrict__ srcidx,
                              const float* __restrict__ vec, float* __restrict__ out, int n)
{
    int i = blockIdx.x * blockDim.x + threadIdx.x;
    if (i >= n) return;
    int g  = seg[i];
    int sr = srcidx ? srcidx[i] : i;
    float w = a[i] / (s[g] + 1e-16f);
    const float* v = vec + (size_t)sr * HH;
    __builtin_prefetch(v, 0, 0);
    float* o = out + (size_t)g * HH;
#pragma unroll
    for (int j = 0; j < HH; ++j) atomicAdd(&o[j], v[j] * w);
}

// ---------------------------------------------------------------------------
// Elementwise pieces
// ---------------------------------------------------------------------------
__global__ void bias_elu_k(float* __restrict__ h, const float* __restrict__ bias, int M)
{
    int t = blockIdx.x * blockDim.x + threadIdx.x;
    if (t >= M * HH) return;
    float v = h[t] + bias[t & 63];
    h[t] = v > 0.f ? v : (__expf(v) - 1.f);
}

// x_new = relu(GRUCell(inp -> gi, hidden -> gh));  gi/gh are [M,192] (no bias yet)
__global__ void gru_combine_k(const float* __restrict__ gi, const float* __restrict__ gh,
                              const float* __restrict__ bi, const float* __restrict__ bh,
                              const float* __restrict__ hprev, float* __restrict__ out, int M)
{
    int t = blockIdx.x * blockDim.x + threadIdx.x;
    if (t >= M * HH) return;
    int m = t >> 6, j = t & 63;
    const float* gim = gi + (size_t)m * 3 * HH;
    const float* ghm = gh + (size_t)m * 3 * HH;
    float ir = gim[j]        + bi[j];
    float iz = gim[HH + j]   + bi[HH + j];
    float in_= gim[2*HH + j] + bi[2*HH + j];
    float hr = ghm[j]        + bh[j];
    float hz = ghm[HH + j]   + bh[HH + j];
    float hn = ghm[2*HH + j] + bh[2*HH + j];
    float r  = 1.f / (1.f + __expf(-(ir + hr)));
    float z  = 1.f / (1.f + __expf(-(iz + hz)));
    float nn = tanhf(in_ + r * hn);
    float v  = (1.f - z) * nn + z * hprev[t];
    out[t] = fmaxf(v, 0.f);
}

__global__ void batch_sum_k(const float* __restrict__ x, const int* __restrict__ batch,
                            float* __restrict__ o)
{
    int t = blockIdx.x * blockDim.x + threadIdx.x;
    if (t >= NN * HH) return;
    atomicAdd(&o[(size_t)batch[t >> 6] * HH + (t & 63)], x[t]);
}

__global__ void relu_k(float* __restrict__ p, int n)
{
    int t = blockIdx.x * blockDim.x + threadIdx.x;
    if (t < n) p[t] = fmaxf(p[t], 0.f);
}

// ---------------------------------------------------------------------------
// Host orchestration
// ---------------------------------------------------------------------------
static inline int cdiv(int a, int b) { return (a + b - 1) / b; }

extern "C" void kernel_launch(void* const* d_in, const int* in_sizes, int n_in,
                              void* d_out, int out_size, void* d_ws, size_t ws_size,
                              hipStream_t stream)
{
    (void)in_sizes; (void)n_in; (void)out_size; (void)ws_size;

    const float* x        = (const float*)d_in[0];
    const int*   eidx     = (const int*)d_in[1];
    const float* eattr    = (const float*)d_in[2];
    const int*   batch    = (const int*)d_in[3];
    const float* lin1_w   = (const float*)d_in[4];
    const float* lin1_b   = (const float*)d_in[5];
    const float* g_lin1_w = (const float*)d_in[6];
    const float* g_att_l  = (const float*)d_in[7];
    const float* g_att_r  = (const float*)d_in[8];
    const float* g_lin2_w = (const float*)d_in[9];
    const float* g_bias   = (const float*)d_in[10];
    const float* gru0_wi  = (const float*)d_in[11];
    const float* gru0_wh  = (const float*)d_in[12];
    const float* gru0_bi  = (const float*)d_in[13];
    const float* gru0_bh  = (const float*)d_in[14];
    const float* conv_lin_w   = (const float*)d_in[15];
    const float* conv_att_src = (const float*)d_in[16];
    const float* conv_att_dst = (const float*)d_in[17];
    const float* conv_bias    = (const float*)d_in[18];
    const float* grul_wi  = (const float*)d_in[19];
    const float* grul_wh  = (const float*)d_in[20];
    const float* grul_bi  = (const float*)d_in[21];
    const float* grul_bh  = (const float*)d_in[22];
    const float* mol_lin_w    = (const float*)d_in[23];
    const float* mol_att_src  = (const float*)d_in[24];
    const float* mol_att_dst  = (const float*)d_in[25];
    const float* mol_bias     = (const float*)d_in[26];
    const float* mgru_wi  = (const float*)d_in[27];
    const float* mgru_wh  = (const float*)d_in[28];
    const float* mgru_bi  = (const float*)d_in[29];
    const float* mgru_bh  = (const float*)d_in[30];
    const float* lin2_w   = (const float*)d_in[31];
    const float* lin2_b   = (const float*)d_in[32];

    const int* src = eidx;
    const int* dst = eidx + EE;

    // ---- workspace arena (~147 MB) ----
    float* f = (float*)d_ws;
    size_t o = 0;
    auto take = [&](size_t n) { float* p = f + o; o += n; return p; };
    float* X0  = take((size_t)NN * HH);
    float* X1  = take((size_t)NN * HH);
    float* TA  = take((size_t)NN * HH);      // tA / hs
    float* Y2  = take((size_t)NN * HH);
    float* Hb  = take((size_t)NN * HH);      // node h / mol h (first G rows)
    float* GI  = take((size_t)NN * 3 * HH);
    float* GH  = take((size_t)NN * 3 * HH);
    float* ER  = take((size_t)EE);           // edge logits / exp values
    float* RD  = take((size_t)NN);           // per-node scalar (src-dot)
    float* AD  = take((size_t)NN);           // per-node scalar (dst-dot)
    float* SM  = take((size_t)NN);           // segment max (uint-keyed, then float)
    float* SS  = take((size_t)NN);           // segment sum
    float* OUTG = take((size_t)GG * HH);
    float* OUT2 = take((size_t)GG * HH);
    float* HD   = take((size_t)GG * HH);
    float* DD   = take((size_t)GG);

    const int B = 256;
    const int nh_grid = cdiv(NN * HH, B);

    // M % 16 == 0 and F % 64 == 0 for every call below.
    auto gemm = [&](const float* A, int M, const float* W, int ldw,
                    const float* bias, float* C, int F) {
        dim3 g(M / 16, F / 64);
        gemm_nt_wmma<<<g, 128, 0, stream>>>(A, W, ldw, bias, C, F);
    };

    // ---- lin1 ----
    lin1_k<<<nh_grid, B, 0, stream>>>(x, lin1_w, lin1_b, X0);

    // ---- GATEConv ----
    gemm(X0, NN, g_lin1_w, HH + EDD, nullptr, TA, HH);   // tA = x0 @ W1a.T
    gemm(X0, NN, g_lin2_w, HH,        nullptr, Y2, HH);  // y2 = x0 @ W2.T
    rowdot_k<<<cdiv(NN, B), B, 0, stream>>>(X0, g_att_r, RD, NN);
    gate_edge_logits_k<<<cdiv(EE, B), B, 0, stream>>>(eattr, src, dst, TA, g_lin1_w,
                                                      g_att_l, RD, ER);
    fill_u32_k<<<cdiv(NN, B), B, 0, stream>>>((unsigned*)SM, ENC_NEG_INF, NN);
    fill_u32_k<<<cdiv(NN, B), B, 0, stream>>>((unsigned*)SS, 0u, NN);
    seg_max_k<<<cdiv(EE, B), B, 0, stream>>>(ER, dst, EE, (unsigned*)SM);
    seg_max_fin_k<<<cdiv(NN, B), B, 0, stream>>>((unsigned*)SM, NN);
    seg_expsum_k<<<cdiv(EE, B), B, 0, stream>>>(ER, dst, EE, SM, SS);
    fill_u32_k<<<nh_grid, B, 0, stream>>>((unsigned*)Hb, 0u, NN * HH);
    seg_scatter_k<<<cdiv(EE, B), B, 0, stream>>>(ER, SS, dst, src, Y2, Hb, EE);
    bias_elu_k<<<nh_grid, B, 0, stream>>>(Hb, g_bias, NN);
    gemm(Hb, NN, gru0_wi, HH, nullptr, GI, 3 * HH);
    gemm(X0, NN, gru0_wh, HH, nullptr, GH, 3 * HH);
    gru_combine_k<<<nh_grid, B, 0, stream>>>(GI, GH, gru0_bi, gru0_bh, X0, X1, NN);
    float* xcur = X1; float* xnext = X0;

    // ---- GATConv layers ----
    for (int l = 0; l < 2; ++l) {
        gemm(xcur, NN, conv_lin_w + (size_t)l * HH * HH, HH, nullptr, TA, HH);
        rowdot_k<<<cdiv(NN, B), B, 0, stream>>>(TA, conv_att_src + (size_t)l * HH, RD, NN);
        rowdot_k<<<cdiv(NN, B), B, 0, stream>>>(TA, conv_att_dst + (size_t)l * HH, AD, NN);
        edge_logits_k<<<cdiv(EE, B), B, 0, stream>>>(RD, AD, src, dst, ER);
        fill_u32_k<<<cdiv(NN, B), B, 0, stream>>>((unsigned*)SM, ENC_NEG_INF, NN);
        fill_u32_k<<<cdiv(NN, B), B, 0, stream>>>((unsigned*)SS, 0u, NN);
        seg_max_k<<<cdiv(EE, B), B, 0, stream>>>(ER, dst, EE, (unsigned*)SM);
        seg_max_fin_k<<<cdiv(NN, B), B, 0, stream>>>((unsigned*)SM, NN);
        seg_expsum_k<<<cdiv(EE, B), B, 0, stream>>>(ER, dst, EE, SM, SS);
        fill_u32_k<<<nh_grid, B, 0, stream>>>((unsigned*)Hb, 0u, NN * HH);
        seg_scatter_k<<<cdiv(EE, B), B, 0, stream>>>(ER, SS, dst, src, TA, Hb, EE);
        bias_elu_k<<<nh_grid, B, 0, stream>>>(Hb, conv_bias + (size_t)l * HH, NN);
        gemm(Hb,   NN, grul_wi + (size_t)l * 3 * HH * HH, HH, nullptr, GI, 3 * HH);
        gemm(xcur, NN, grul_wh + (size_t)l * 3 * HH * HH, HH, nullptr, GH, 3 * HH);
        gru_combine_k<<<nh_grid, B, 0, stream>>>(GI, GH, grul_bi + (size_t)l * 3 * HH,
                                                 grul_bh + (size_t)l * 3 * HH,
                                                 xcur, xnext, NN);
        float* t = xcur; xcur = xnext; xnext = t;
    }

    // ---- molecule-level attention readout ----
    fill_u32_k<<<cdiv(GG * HH, B), B, 0, stream>>>((unsigned*)OUTG, 0u, GG * HH);
    batch_sum_k<<<nh_grid, B, 0, stream>>>(xcur, batch, OUTG);
    relu_k<<<cdiv(GG * HH, B), B, 0, stream>>>(OUTG, GG * HH);
    float* outcur = OUTG; float* outnext = OUT2;

    for (int t = 0; t < 2; ++t) {
        gemm(xcur,   NN, mol_lin_w, HH, nullptr, TA, HH);  // hs (atoms)
        gemm(outcur, GG, mol_lin_w, HH, nullptr, HD, HH);  // hd (mols)
        rowdot_k<<<cdiv(NN, B), B, 0, stream>>>(TA, mol_att_src, RD, NN);
        rowdot_k<<<cdiv(GG, B), B, 0, stream>>>(HD, mol_att_dst, DD, GG);
        mol_logits_k<<<cdiv(NN, B), B, 0, stream>>>(RD, DD, batch, ER);
        fill_u32_k<<<cdiv(GG, B), B, 0, stream>>>((unsigned*)SM, ENC_NEG_INF, GG);
        fill_u32_k<<<cdiv(GG, B), B, 0, stream>>>((unsigned*)SS, 0u, GG);
        seg_max_k<<<cdiv(NN, B), B, 0, stream>>>(ER, batch, NN, (unsigned*)SM);
        seg_max_fin_k<<<cdiv(GG, B), B, 0, stream>>>((unsigned*)SM, GG);
        seg_expsum_k<<<cdiv(NN, B), B, 0, stream>>>(ER, batch, NN, SM, SS);
        fill_u32_k<<<cdiv(GG * HH, B), B, 0, stream>>>((unsigned*)Hb, 0u, GG * HH);
        seg_scatter_k<<<cdiv(NN, B), B, 0, stream>>>(ER, SS, batch, nullptr, TA, Hb, NN);
        bias_elu_k<<<cdiv(GG * HH, B), B, 0, stream>>>(Hb, mol_bias, GG);
        gemm(Hb,     GG, mgru_wi, HH, nullptr, GI, 3 * HH);
        gemm(outcur, GG, mgru_wh, HH, nullptr, GH, 3 * HH);
        gru_combine_k<<<cdiv(GG * HH, B), B, 0, stream>>>(GI, GH, mgru_bi, mgru_bh,
                                                          outcur, outnext, GG);
        float* tp = outcur; outcur = outnext; outnext = tp;
    }

    // ---- lin2 -> d_out  [1, G, OUT_C] ----
    gemm(outcur, GG, lin2_w, HH, lin2_b, (float*)d_out, OUT_CC);
}
